// SchNetEncoder_67688684585228
// MI455X (gfx1250) — matrix operand order repaired and checked by reading, not compiled, not run on hardware
//

#include <hip/hip_runtime.h>
#include <hip/hip_bf16.h>
#include <math.h>

// ---------------------------------------------------------------------------
// SchNet encoder for MI455X (gfx1250, wave32, WMMA).
// All GEMM-like work routed through v_wmma_f32_16x16x32_f16.
// ---------------------------------------------------------------------------

typedef __attribute__((ext_vector_type(16))) _Float16 v16h;
typedef __attribute__((ext_vector_type(8)))  float    v8f;

#define NNODES 50000
#define NEDGES 1600000
#define HDIM   128
#define NGAUSS 50
#define NLAYER 5
#define CUTOFF 4.5f
#define LOG2F_ 0.69314718055994531f

// ---- WMMA fragment builders (ISA 7.12.2 layouts, wave32) -------------------
// A matrix 16x32 f16: lane l -> row m=l&15, half=l>>4.
//   VGPR v (pair 2v,2v+1): k = kbase + (v<4?0:16) + half*8 + (v&3)*2, +1
__device__ __forceinline__ v16h load_a_frag(const _Float16* Af, int lda,
                                            int m, int half, int kbase) {
  v16h a;
#pragma unroll
  for (int v = 0; v < 8; ++v) {
    int k = kbase + ((v < 4) ? 0 : 16) + half * 8 + (v & 3) * 2;
    a[2 * v]     = Af[m * lda + k];
    a[2 * v + 1] = Af[m * lda + k + 1];
  }
  return a;
}

// B matrix 32x16 f16: lane l -> col n=l&15, half=l>>4.
//   VGPR v (pair): k = kbase + (v<4?0:16) + half*8 + (v&3)*2, +1.
// Bt is stored transposed in LDS: Bt[col*ldb + k]  (K contiguous -> b32 reads)
__device__ __forceinline__ v16h load_b_frag(const _Float16* Bt, int ldb,
                                            int col, int half, int kbase) {
  v16h b;
#pragma unroll
  for (int v = 0; v < 8; ++v) {
    int k = kbase + ((v < 4) ? 0 : 16) + half * 8 + (v & 3) * 2;
    b[2 * v]     = Bt[col * ldb + k];
    b[2 * v + 1] = Bt[col * ldb + k + 1];
  }
  return b;
}

// ---- embedding gather ------------------------------------------------------
__global__ __launch_bounds__(256) void k_embed(const int* __restrict__ an,
                                               const float* __restrict__ emb,
                                               float* __restrict__ x) {
  int i   = blockIdx.x * 256 + threadIdx.x;   // over N*H, exact multiple
  int row = i >> 7;
  int col = i & 127;
  x[i] = emb[an[row] * HDIM + col];
}

// ---- zero scratch ----------------------------------------------------------
__global__ __launch_bounds__(256) void k_zero(float* __restrict__ p) {
  p[blockIdx.x * 256 + threadIdx.x] = 0.0f;
}

// ---- node GEMM: out = in @ W + b, templated epilogue -----------------------
// 625 blocks x 160 threads (5 waves). Block stages 80 rows of A (f16) and the
// whole 128x128 W^T (f16) once; each wave owns a 16-row strip and computes all
// 8 column tiles (32 WMMAs/wave). 625*80 == 50000 exactly -> no guards, EXEC
// all-ones at every WMMA.
// MODE 0: store   MODE 1: shifted softplus   MODE 2: out = res + v (residual)
template <int MODE>
__global__ __launch_bounds__(160) void k_gemm_node(
    const float* __restrict__ in, const float* __restrict__ Wg,
    const float* __restrict__ bias, const float* __restrict__ res,
    float* __restrict__ out) {
  __shared__ _Float16 Af[80 * 128];    // 20 KB
  __shared__ _Float16 Bt[128 * 128];   // 32 KB (W transposed, f16)
  __shared__ float    bs[128];

  const int tid  = threadIdx.x;
  const int rowB = blockIdx.x * 80;    // first row of this block

  for (int i = tid; i < 80 * 128; i += 160) {
    int r = i >> 7, c = i & 127;
    Af[i] = (_Float16)in[(rowB + r) * HDIM + c];
  }
  for (int i = tid; i < 128 * 128; i += 160) {
    int k = i >> 7, c = i & 127;
    Bt[c * 128 + k] = (_Float16)Wg[i];
  }
  if (tid < 128) bs[tid] = bias[tid];
  __syncthreads();

  const int lane = tid & 31;
  const int wave = tid >> 5;           // 0..4 : 16-row strip within block
  const int m    = lane & 15;
  const int half = lane >> 4;
  const int row0 = rowB + wave * 16;

  v8f acc[8];
#pragma unroll
  for (int t = 0; t < 8; ++t) acc[t] = {};

#pragma unroll
  for (int ks = 0; ks < 4; ++ks) {
    v16h a = load_a_frag(Af + wave * 16 * 128, 128, m, half, ks * 32);
#pragma unroll
    for (int t = 0; t < 8; ++t) {
      v16h b = load_b_frag(Bt, 128, t * 16 + m, half, ks * 32);
      acc[t] = __builtin_amdgcn_wmma_f32_16x16x32_f16(
          false, a, false, b, (short)0, acc[t], false, false);
    }
  }

#pragma unroll
  for (int t = 0; t < 8; ++t) {
    const int   col  = t * 16 + m;
    const float bcol = bs[col];
#pragma unroll
    for (int r = 0; r < 8; ++r) {
      int   row = row0 + r + 8 * half;   // C/D layout: row = r + 8*half
      float v   = acc[t][r] + bcol;
      if (MODE == 1) {
        float sp = (v > 20.0f) ? v : log1pf(__expf(v));
        v = sp - LOG2F_;
      } else if (MODE == 2) {
        v += res[row * HDIM + col];
      }
      out[row * HDIM + col] = v;
    }
  }
}

// ---- fused edge phase ------------------------------------------------------
// For 16 edges per wave: A = gaussian-smearing rows (K padded 50->64, built
// straight into the WMMA A layout in registers), B = Wn^T (f16, LDS).
// acc = gauss@Wn + bn via 16 WMMAs; epilogue applies cosine cutoff, gathers
// h[src] (L2-resident) and scatter-adds into agg[dst] with f32 atomics.
__global__ __launch_bounds__(256) void k_edge(
    const int* __restrict__ src, const int* __restrict__ dst,
    const float* __restrict__ attr, const float* __restrict__ Wn,
    const float* __restrict__ bn, const float* __restrict__ h,
    float* __restrict__ agg) {
  __shared__ _Float16 Bt[128 * 64];    // 16 KB : Wn transposed, K padded to 64
  __shared__ float    bns[128];

  const int tid = threadIdx.x;
  {  // prefetch this block's edge streams into cache (global_prefetch_b8)
    long eb = (long)blockIdx.x * 128;
    __builtin_prefetch(src + eb, 0, 1);
    __builtin_prefetch(dst + eb, 0, 1);
    __builtin_prefetch(attr + eb, 0, 1);
  }
  for (int i = tid; i < NGAUSS * 128; i += 256) {
    int k = i >> 7, c = i & 127;
    Bt[c * 64 + k] = (_Float16)Wn[i];
  }
  for (int i = tid; i < (64 - NGAUSS) * 128; i += 256) {
    int k = NGAUSS + (i >> 7), c = i & 127;
    Bt[c * 64 + k] = (_Float16)0.0f;
  }
  if (tid < 128) bns[tid] = bn[tid];
  __syncthreads();

  const int  lane = tid & 31;
  const int  wave = tid >> 5;
  const int  m    = lane & 15;
  const int  half = lane >> 4;
  const long e0   = (long)blockIdx.x * 128 + wave * 16;

  const float step  = CUTOFF / (float)(NGAUSS - 1);
  const float coeff = -0.5f / (step * step);
  const float d_m   = attr[e0 + m];   // edge owned by this lane's A rows

  v8f acc[8];
#pragma unroll
  for (int t = 0; t < 8; ++t) {
    float bv = bns[t * 16 + m];
#pragma unroll
    for (int r = 0; r < 8; ++r) acc[t][r] = bv;
  }

#pragma unroll
  for (int ks = 0; ks < 2; ++ks) {
    v16h a;
#pragma unroll
    for (int v = 0; v < 8; ++v) {
      int kb = ks * 32 + ((v < 4) ? 0 : 16) + half * 8 + (v & 3) * 2;
#pragma unroll
      for (int j = 0; j < 2; ++j) {
        int   k = kb + j;
        float g = 0.0f;
        if (k < NGAUSS) {
          float t0 = d_m - (float)k * step;
          g = __expf(coeff * t0 * t0);
        }
        a[2 * v + j] = (_Float16)g;
      }
    }
#pragma unroll
    for (int t = 0; t < 8; ++t) {
      v16h b = load_b_frag(Bt, 64, t * 16 + m, half, ks * 32);
      acc[t] = __builtin_amdgcn_wmma_f32_16x16x32_f16(
          false, a, false, b, (short)0, acc[t], false, false);
    }
  }

  const float PIC = 3.14159265358979f / CUTOFF;
#pragma unroll
  for (int r = 0; r < 8; ++r) {
    long  e  = e0 + r + 8 * half;       // C/D layout: row = r + 8*half
    int   s  = src[e];
    int   dd = dst[e];
    float de = attr[e];
    float Ce = 0.5f * (__cosf(de * PIC) + 1.0f);
    const float* hrow = h + (long)s * HDIM;
    float*       arow = agg + (long)dd * HDIM;
#pragma unroll
    for (int t = 0; t < 8; ++t) {
      int   col = t * 16 + m;           // C/D layout: col = lane&15
      float val = acc[t][r] * Ce * hrow[col];
      atomicAdd(&arow[col], val);
    }
  }
}

// ---------------------------------------------------------------------------
extern "C" void kernel_launch(void* const* d_in, const int* in_sizes, int n_in,
                              void* d_out, int out_size, void* d_ws,
                              size_t ws_size, hipStream_t stream) {
  const int*   an   = (const int*)d_in[0];
  const int*   ei   = (const int*)d_in[1];        // [2, E]
  const float* attr = (const float*)d_in[2];
  const float* emb  = (const float*)d_in[3];
  const float* Wi   = (const float*)d_in[4];
  const float* bi   = (const float*)d_in[5];
  const float* Wn   = (const float*)d_in[6];
  const float* bn   = (const float*)d_in[7];
  const float* Wo   = (const float*)d_in[8];
  const float* bo   = (const float*)d_in[9];
  const float* Wl   = (const float*)d_in[10];
  const float* bl   = (const float*)d_in[11];

  const int* src = ei;
  const int* dst = ei + NEDGES;

  float* x   = (float*)d_out;                 // [N,H], updated in place
  float* h   = (float*)d_ws;                  // [N,H]
  float* agg = h + (size_t)NNODES * HDIM;     // [N,H]

  const int NH_BLK   = (NNODES * HDIM) / 256; // 25000, exact
  const int GEMM_BLK = NNODES / 80;           // 625, exact
  const int EDGE_BLK = NEDGES / 128;          // 12500, exact

  k_embed<<<NH_BLK, 256, 0, stream>>>(an, emb, x);

  for (int l = 0; l < NLAYER; ++l) {
    const float* Wi_l = Wi + (size_t)l * HDIM * HDIM;
    const float* Wn_l = Wn + (size_t)l * NGAUSS * HDIM;
    const float* Wo_l = Wo + (size_t)l * HDIM * HDIM;
    const float* Wl_l = Wl + (size_t)l * HDIM * HDIM;

    // h = x @ Wi + bi
    k_gemm_node<0><<<GEMM_BLK, 160, 0, stream>>>(x, Wi_l, bi + l * HDIM,
                                                 nullptr, h);
    // agg = segment_sum(h[src] * ((gauss@Wn + bn) * C), dst)
    k_zero<<<NH_BLK, 256, 0, stream>>>(agg);
    k_edge<<<EDGE_BLK, 256, 0, stream>>>(src, dst, attr, Wn_l, bn + l * HDIM,
                                         h, agg);
    // h = softplus(agg @ Wo + bo) - log2     (reuse h as the intermediate)
    k_gemm_node<1><<<GEMM_BLK, 160, 0, stream>>>(agg, Wo_l, bo + l * HDIM,
                                                 nullptr, h);
    // x = x + (h @ Wl + bl)
    k_gemm_node<2><<<GEMM_BLK, 160, 0, stream>>>(h, Wl_l, bl + l * HDIM,
                                                 x, x);
  }
}
